// DeformingPlateModel_10522669875484
// MI455X (gfx1250) — matrix operand order repaired and checked by reading, not compiled, not run on hardware
//
#include <hip/hip_runtime.h>
#include <stdint.h>
#include <stddef.h>

// ---------------------------------------------------------------------------
// MeshGraphNets (DeformingPlate) on gfx1250.
// All GEMMs run on v_wmma_f32_16x16x32_bf16 (wave32, 16x16 tiles).
// Weights are staged through LDS once per workgroup (ds_load_b128 in the
// hot loop); activations stream from global memory (global_load_b128).
// ---------------------------------------------------------------------------

#define LATENT    128
#define MP_STEPS  15
#define NTYPES    9

typedef __attribute__((ext_vector_type(16))) __bf16          v16bf;
typedef __attribute__((ext_vector_type(8)))  float           v8f;
typedef __attribute__((ext_vector_type(8)))  unsigned short  us8;

union FragAB { v16bf v; us8 h[2]; };

// Force global (addrspace 1) memory ops so the backend emits
// global_load_b128 / global_store_* instead of flat_* (no DScnt coupling).
typedef us8            __attribute__((address_space(1))) us8_g;
typedef float          __attribute__((address_space(1))) float_g;
typedef unsigned short __attribute__((address_space(1))) ushort_g;
typedef int            __attribute__((address_space(1))) int_g;

__device__ __forceinline__ us8 gload8(const unsigned short* p) {
  return *(const us8_g*)(unsigned long long)(size_t)p;
}
__device__ __forceinline__ int gloadi(const int* p) {
  return *(const int_g*)(unsigned long long)(size_t)p;
}
__device__ __forceinline__ float gloadf(const float* p) {
  return *(const float_g*)(unsigned long long)(size_t)p;
}
__device__ __forceinline__ void gstoref(float* p, float v) {
  *(float_g*)(unsigned long long)(size_t)p = v;
}
__device__ __forceinline__ void gstoreu(unsigned short* p, unsigned short v) {
  *(ushort_g*)(unsigned long long)(size_t)p = v;
}

__device__ __forceinline__ unsigned short f2bf(float f) {
  unsigned int u = __float_as_uint(f);
  unsigned int r = u + 0x7fffu + ((u >> 16) & 1u);   // round-to-nearest-even
  return (unsigned short)(r >> 16);
}

// ---------------------------------------------------------------------------
// Weight repack: f32 [Kin x Ncols] row-major -> bf16 WMMA B-operand fragments.
// B layout (this code's contiguous k order, same as A): lane l -> col = t*16 + (l&15); lanes>=16 hold K 16..31
// of the 32-K chunk; element j -> k = kc*32 + (l>>4)*16 + j.
// ---------------------------------------------------------------------------
__global__ void pack_w_kernel(const float* __restrict__ W,
                              unsigned short* __restrict__ Wf,
                              int Kin, int Ncols, int Ktot, int NT) {
  int tid = blockIdx.x * blockDim.x + threadIdx.x;
  int total = (Ktot >> 5) * NT * 32;
  if (tid >= total) return;
  int lane = tid & 31;
  int t    = (tid >> 5) % NT;
  int kc   = (tid >> 5) / NT;
  int col  = t * 16 + (lane & 15);
  int kb   = kc * 32 + ((lane >> 4) << 4);
  unsigned short* dst = Wf + (size_t)tid * 16;
#pragma unroll
  for (int j = 0; j < 16; ++j) {
    int k = kb + j;
    float v = (k < Kin && col < Ncols) ? gloadf(W + (size_t)k * Ncols + col) : 0.0f;
    gstoreu(dst + j, f2bf(v));
  }
}

__global__ void cvt_bf16_kernel(const float* __restrict__ src,
                                unsigned short* __restrict__ dst, long n) {
  long i = (long)blockIdx.x * blockDim.x + threadIdx.x;
  if (i < n) gstoreu(dst + i, f2bf(gloadf(src + i)));
}

// ---------------------------------------------------------------------------
// Feature builders (normalized, bf16, K padded to 32)
// ---------------------------------------------------------------------------
__global__ void node_feat_kernel(const int* __restrict__ node_type,
                                 const float* __restrict__ known_vel,
                                 const float* __restrict__ mean,
                                 const float* __restrict__ stdv,
                                 unsigned short* __restrict__ xn, int N) {
  int i = blockIdx.x * blockDim.x + threadIdx.x;
  if (i >= N) return;
  float f[12];
#pragma unroll
  for (int j = 0; j < 12; ++j) f[j] = 0.0f;
  f[0] = gloadf(known_vel + i * 3 + 0);
  f[1] = gloadf(known_vel + i * 3 + 1);
  f[2] = gloadf(known_vel + i * 3 + 2);
  int t = gloadi(node_type + i);
  if (t >= 0 && t < NTYPES) f[3 + t] = 1.0f;
  unsigned short* o = xn + (size_t)i * 32;
#pragma unroll
  for (int j = 0; j < 12; ++j) gstoreu(o + j, f2bf((f[j] - mean[j]) / stdv[j]));
#pragma unroll
  for (int j = 12; j < 32; ++j) gstoreu(o + j, 0);
}

__global__ void mesh_feat_kernel(const int* __restrict__ srcs,
                                 const int* __restrict__ dsts,
                                 const float* __restrict__ mesh_pos,
                                 const float* __restrict__ world_pos,
                                 const float* __restrict__ mean,
                                 const float* __restrict__ stdv,
                                 unsigned short* __restrict__ xm, int E) {
  int e = blockIdx.x * blockDim.x + threadIdx.x;
  if (e >= E) return;
  int s = gloadi(srcs + e), d = gloadi(dsts + e);
  float rm[3], rw[3], nm = 0.f, nw = 0.f;
#pragma unroll
  for (int j = 0; j < 3; ++j) {
    rm[j] = gloadf(mesh_pos + s * 3 + j) - gloadf(mesh_pos + d * 3 + j);
    rw[j] = gloadf(world_pos + s * 3 + j) - gloadf(mesh_pos + d * 3 + j);
    nm += rm[j] * rm[j];
    nw += rw[j] * rw[j];
  }
  nm = sqrtf(nm); nw = sqrtf(nw);
  float f[8] = {rm[0], rm[1], rm[2], nm, rw[0], rw[1], rw[2], nw};
  unsigned short* o = xm + (size_t)e * 32;
#pragma unroll
  for (int j = 0; j < 8; ++j) gstoreu(o + j, f2bf((f[j] - mean[j]) / stdv[j]));
#pragma unroll
  for (int j = 8; j < 32; ++j) gstoreu(o + j, 0);
}

__global__ void world_feat_kernel(const int* __restrict__ wsrcs,
                                  const int* __restrict__ wdsts,
                                  const float* __restrict__ world_pos,
                                  const float* __restrict__ mean,
                                  const float* __restrict__ stdv,
                                  unsigned short* __restrict__ xw, int E) {
  int e = blockIdx.x * blockDim.x + threadIdx.x;
  if (e >= E) return;
  int s = gloadi(wsrcs + e), d = gloadi(wdsts + e);
  float r[3], n = 0.f;
#pragma unroll
  for (int j = 0; j < 3; ++j) {
    r[j] = gloadf(world_pos + s * 3 + j) - gloadf(world_pos + d * 3 + j);
    n += r[j] * r[j];
  }
  n = sqrtf(n);
  float f[4] = {r[0], r[1], r[2], n};
  unsigned short* o = xw + (size_t)e * 32;
#pragma unroll
  for (int j = 0; j < 4; ++j) gstoreu(o + j, f2bf((f[j] - mean[j]) / stdv[j]));
#pragma unroll
  for (int j = 4; j < 32; ++j) gstoreu(o + j, 0);
}

// ---------------------------------------------------------------------------
// Generic WMMA GEMM + fused epilogue.
//   Y[M x NT*16] = epilogue( concat_k(X0[i0],X1[i1],X2[i2]) @ Wf + bias )
// One wave computes a 16 x (NT*16) tile; block = 8 waves = 128 rows.
// Packed weights are staged to LDS once per workgroup (dyn-shared =
// kcTot*NT*1024 bytes, <= 96KB), so the hot loop is ds_load_b128 for B
// fragments + one global A gather per 32-K chunk.
// MODE 0: ReLU -> bf16 out (hidden layers)
// MODE 1: LayerNorm -> f32 out + bf16 mirror (encoders)
// MODE 2: LayerNorm -> residual += , f32 + bf16 mirror (node update)
// MODE 3: MODE 2 + atomic scatter-add of pre-residual value into agg (edges)
// MODE 4: y*scale + shift, first 3 cols -> f32 (decoder output)
// ---------------------------------------------------------------------------
template <int MODE, int NT>
__global__ void __launch_bounds__(256)
gemm_mlp(int M, int Kseg, int nseg,
         const unsigned short* __restrict__ X0,
         const unsigned short* __restrict__ X1,
         const unsigned short* __restrict__ X2,
         const int* __restrict__ idx0,
         const int* __restrict__ idx1,
         const int* __restrict__ idx2,
         const unsigned short* __restrict__ Wf,
         const float* __restrict__ bias, int biasN,
         const float* __restrict__ ln_g, const float* __restrict__ ln_b,
         float* __restrict__ outF, unsigned short* __restrict__ outB,
         const int* __restrict__ scatIdx, float* __restrict__ agg,
         const float* __restrict__ oscale, const float* __restrict__ oshift) {
  extern __shared__ unsigned short smem[];

  const int kcTot = (nseg * Kseg) >> 5;

  // ---- cooperative stage of packed weights into LDS (all waves) ----
  {
    const int chunks = kcTot * NT * 64;  // 16-byte chunks
    for (int c = threadIdx.x; c < chunks; c += 256)
      *(us8*)(smem + (size_t)c * 8) = gload8(Wf + (size_t)c * 8);
  }
  __syncthreads();

  const int lane = threadIdx.x & 31;
  const int wave = threadIdx.x >> 5;
  const int rowbase = blockIdx.x * 128 + wave * 16;
  if (rowbase >= M) return;

  const int colw = lane & 15;
  const int half = lane >> 4;

  // A-operand row pointers (optionally gathered through edge index arrays)
  const int rA = rowbase + colw;          // A layout: lane&15 selects the row
  const int rAc = rA < M ? rA : (M - 1);
  const unsigned short* xrow[3] = {nullptr, nullptr, nullptr};
  {
    const unsigned short* Xs[3] = {X0, X1, X2};
    const int* Is[3] = {idx0, idx1, idx2};
    for (int s = 0; s < nseg; ++s) {
      int sr = Is[s] ? gloadi(Is[s] + rAc) : rAc;
      xrow[s] = Xs[s] + (size_t)sr * Kseg;
    }
  }

  // Accumulators initialized with bias (bias depends on column only)
  v8f acc[NT];
#pragma unroll
  for (int t = 0; t < NT; ++t) {
    int c = t * 16 + colw;
    float bv = (bias != nullptr && c < biasN) ? gloadf(bias + c) : 0.0f;
#pragma unroll
    for (int j = 0; j < 8; ++j) acc[t][j] = bv;
  }

  // K loop, 32 per iteration. A layout (contiguous k order, same as packed B):
  // lanes 0-15 hold K {0..15}, lanes 16-31 hold K {16..31}.
  const int kbA = half << 4;
  int seg = 0, ks = 0;
  for (int kc = 0; kc < kcTot; ++kc) {
    const unsigned short* ap = xrow[seg] + ks + kbA;
    const unsigned short* wp = smem + ((size_t)kc * NT * 32 + lane) * 16;

    FragAB a;
    a.h[0] = gload8(ap);
    a.h[1] = gload8(ap + 8);

    FragAB b[NT];
#pragma unroll
    for (int t = 0; t < NT; ++t) {
      const unsigned short* wpt = wp + (size_t)t * 512;
      b[t].h[0] = *(const us8*)(wpt);        // ds_load_b128
      b[t].h[1] = *(const us8*)(wpt + 8);
    }
#pragma unroll
    for (int t = 0; t < NT; ++t) {
      acc[t] = __builtin_amdgcn_wmma_f32_16x16x32_bf16(
          false, a.v, false, b[t].v, (short)0, acc[t], false, false);
    }
    ks += 32;
    if (ks == Kseg) { ks = 0; ++seg; }
  }

  // Epilogues. C/D layout: lanes 0-15: VGPR j -> row j; lanes 16-31: row 8+j;
  // column = t*16 + (lane&15).
  const int rhalf = half << 3;

  if (MODE == 0) {  // ReLU -> bf16
#pragma unroll
    for (int j = 0; j < 8; ++j) {
      int gr = rowbase + rhalf + j;
      if (gr < M) {
        unsigned short* orow = outB + (size_t)gr * (NT * 16);
#pragma unroll
        for (int t = 0; t < NT; ++t) {
          float y = acc[t][j];
          gstoreu(orow + t * 16 + colw, f2bf(y > 0.0f ? y : 0.0f));
        }
      }
    }
  } else if (MODE == 4) {  // decoder: out = y*std + mean, 3 cols
#pragma unroll
    for (int j = 0; j < 8; ++j) {
      int gr = rowbase + rhalf + j;
      if (gr < M && colw < 3) {
        gstoref(outF + (size_t)gr * 3 + colw,
                acc[0][j] * gloadf(oscale + colw) + gloadf(oshift + colw));
      }
    }
  } else {  // LayerNorm modes (NT == 8, n = 128)
    float gln[NT], bln[NT];
#pragma unroll
    for (int t = 0; t < NT; ++t) {
      gln[t] = gloadf(ln_g + t * 16 + colw);
      bln[t] = gloadf(ln_b + t * 16 + colw);
    }
#pragma unroll
    for (int j = 0; j < 8; ++j) {
      float s1 = 0.f, s2 = 0.f;
#pragma unroll
      for (int t = 0; t < NT; ++t) {
        float x = acc[t][j];
        s1 += x; s2 += x * x;
      }
      // row lives in one 16-lane half of the wave32: xor masks 1,2,4,8
#pragma unroll
      for (int m = 1; m < 16; m <<= 1) {
        s1 += __shfl_xor(s1, m, 32);
        s2 += __shfl_xor(s2, m, 32);
      }
      const float inv = 1.0f / (NT * 16);
      float mu = s1 * inv;
      float var = s2 * inv - mu * mu;
      float rstd = rsqrtf(var + 1e-5f);
      int gr = rowbase + rhalf + j;
      if (gr >= M) continue;
      int sd = (MODE == 3) ? gloadi(scatIdx + gr) : 0;
      size_t ro = (size_t)gr * LATENT;
#pragma unroll
      for (int t = 0; t < NT; ++t) {
        int c = t * 16 + colw;
        float y = (acc[t][j] - mu) * rstd * gln[t] + bln[t];
        if (MODE == 1) {
          gstoref(outF + ro + c, y);
          gstoreu(outB + ro + c, f2bf(y));
        } else {
          if (MODE == 3) atomicAdd(&agg[(size_t)sd * LATENT + c], y);
          float nv = gloadf(outF + ro + c) + y;  // residual
          gstoref(outF + ro + c, nv);
          gstoreu(outB + ro + c, f2bf(nv));
        }
      }
    }
  }
}

// ---------------------------------------------------------------------------
// Host side
// ---------------------------------------------------------------------------
extern "C" void kernel_launch(void* const* d_in, const int* in_sizes, int n_in,
                              void* d_out, int out_size, void* d_ws, size_t ws_size,
                              hipStream_t stream) {
  (void)n_in; (void)out_size; (void)ws_size;

  const int*   node_type = (const int*)d_in[1];
  const float* mesh_pos  = (const float*)d_in[2];
  const float* world_pos = (const float*)d_in[3];
  const float* known_vel = (const float*)d_in[4];
  const int*   srcs      = (const int*)d_in[5];
  const int*   dsts      = (const int*)d_in[6];
  const int*   wsrcs     = (const int*)d_in[7];
  const int*   wdsts     = (const int*)d_in[8];
  const int N  = in_sizes[1];
  const int E0 = in_sizes[5];
  const int EW = in_sizes[7];

  // Params flattened as JAX pytree leaves (dict keys sorted alphabetically):
  // blocks[15]{e0,e1,n}{Ws0,Ws1,Ws2,bs0,bs1,bs2,ln_b,ln_g}, decoder{W*3,b*3},
  // mesh_enc(8), mesh_norm(2), node_enc(8), node_norm(2), out_norm(2),
  // world_enc(8), world_norm(2).
  auto P = [&](int i) -> const float* { return (const float*)d_in[i]; };
  auto BLK = [&](int i, int m, int l) -> const float* {
    return (const float*)d_in[9 + i * 24 + m * 8 + l];
  };
  const int DEC = 369, MESHE = 375, MESHN = 383, NODEE = 385, NODEN = 393,
            OUTN = 395, WORLDE = 397, WORLDN = 405;

  // ---- deterministic workspace carving ----
  char* ws = (char*)d_ws;
  size_t off = 0;
  auto alloc = [&](size_t bytes) -> void* {
    off = (off + 255) & ~(size_t)255;
    void* p = ws + off;
    off += bytes;
    return p;
  };

  auto packLayer = [&](const float* W, int Kin, int Ncols, int Ktot,
                       int NT) -> unsigned short* {
    size_t elems = (size_t)(Ktot >> 5) * NT * 512;
    unsigned short* Wf = (unsigned short*)alloc(elems * 2);
    int total = (Ktot >> 5) * NT * 32;
    pack_w_kernel<<<(total + 255) / 256, 256, 0, stream>>>(W, Wf, Kin, Ncols,
                                                           Ktot, NT);
    return Wf;
  };

  // packed weights
  unsigned short* neW[3] = {packLayer(P(NODEE + 0), 12, 128, 32, 8),
                            packLayer(P(NODEE + 1), 128, 128, 128, 8),
                            packLayer(P(NODEE + 2), 128, 128, 128, 8)};
  unsigned short* meW[3] = {packLayer(P(MESHE + 0), 8, 128, 32, 8),
                            packLayer(P(MESHE + 1), 128, 128, 128, 8),
                            packLayer(P(MESHE + 2), 128, 128, 128, 8)};
  unsigned short* weW[3] = {packLayer(P(WORLDE + 0), 4, 128, 32, 8),
                            packLayer(P(WORLDE + 1), 128, 128, 128, 8),
                            packLayer(P(WORLDE + 2), 128, 128, 128, 8)};
  unsigned short* dcW[3] = {packLayer(P(DEC + 0), 128, 128, 128, 8),
                            packLayer(P(DEC + 1), 128, 128, 128, 8),
                            packLayer(P(DEC + 2), 128, 3, 128, 1)};
  unsigned short* bW[MP_STEPS][3][3];
  for (int i = 0; i < MP_STEPS; ++i)
    for (int m = 0; m < 3; ++m) {
      bW[i][m][0] = packLayer(BLK(i, m, 0), 384, 128, 384, 8);
      bW[i][m][1] = packLayer(BLK(i, m, 1), 128, 128, 128, 8);
      bW[i][m][2] = packLayer(BLK(i, m, 2), 128, 128, 128, 8);
    }

  // activations
  const long maxM = (long)(E0 > N ? (E0 > EW ? E0 : EW) : (N > EW ? N : EW));
  unsigned short* xn = (unsigned short*)alloc((size_t)N * 32 * 2);
  unsigned short* xm = (unsigned short*)alloc((size_t)E0 * 32 * 2);
  unsigned short* xw = (unsigned short*)alloc((size_t)(EW > 0 ? EW : 1) * 32 * 2);
  float*          vF = (float*)alloc((size_t)N * LATENT * 4);
  unsigned short* vB = (unsigned short*)alloc((size_t)N * LATENT * 2);
  float*          e0F = (float*)alloc((size_t)E0 * LATENT * 4);
  unsigned short* e0B = (unsigned short*)alloc((size_t)E0 * LATENT * 2);
  float*          e1F = (float*)alloc((size_t)(EW > 0 ? EW : 1) * LATENT * 4);
  unsigned short* e1B = (unsigned short*)alloc((size_t)(EW > 0 ? EW : 1) * LATENT * 2);
  float*          ag0 = (float*)alloc((size_t)N * LATENT * 4);
  float*          ag1 = (float*)alloc((size_t)N * LATENT * 4);
  unsigned short* ag0B = (unsigned short*)alloc((size_t)N * LATENT * 2);
  unsigned short* ag1B = (unsigned short*)alloc((size_t)N * LATENT * 2);
  unsigned short* h0 = (unsigned short*)alloc((size_t)maxM * LATENT * 2);
  unsigned short* h1 = (unsigned short*)alloc((size_t)maxM * LATENT * 2);

  auto grid = [](int M) { return dim3((unsigned)((M + 127) / 128)); };
  // dynamic LDS bytes for a gemm launch: kcTot * NT * 1024
  auto shmem = [](int Kseg, int nseg, int NT) -> size_t {
    return (size_t)((nseg * Kseg) >> 5) * NT * 1024;
  };

  auto gemmRelu = [&](int M, int Kseg, int nseg, const unsigned short* X0,
                      const unsigned short* X1, const unsigned short* X2,
                      const int* i0, const int* i1, const int* i2,
                      const unsigned short* Wf, const float* bias,
                      unsigned short* outB) {
    gemm_mlp<0, 8><<<grid(M), 256, shmem(Kseg, nseg, 8), stream>>>(
        M, Kseg, nseg, X0, X1, X2, i0, i1, i2, Wf, bias, 128, nullptr, nullptr,
        nullptr, outB, nullptr, nullptr, nullptr, nullptr);
  };
  auto cvt = [&](const float* s, unsigned short* d, long n) {
    cvt_bf16_kernel<<<(unsigned)((n + 255) / 256), 256, 0, stream>>>(s, d, n);
  };

  // ---- encoders ----
  node_feat_kernel<<<(N + 255) / 256, 256, 0, stream>>>(node_type, known_vel,
                                                        P(NODEN), P(NODEN + 1), xn, N);
  mesh_feat_kernel<<<(E0 + 255) / 256, 256, 0, stream>>>(
      srcs, dsts, mesh_pos, world_pos, P(MESHN), P(MESHN + 1), xm, E0);
  if (EW > 0)
    world_feat_kernel<<<(EW + 255) / 256, 256, 0, stream>>>(
        wsrcs, wdsts, world_pos, P(WORLDN), P(WORLDN + 1), xw, EW);

  // node encoder -> v
  gemmRelu(N, 32, 1, xn, nullptr, nullptr, nullptr, nullptr, nullptr, neW[0], P(NODEE + 3), h0);
  gemmRelu(N, 128, 1, h0, nullptr, nullptr, nullptr, nullptr, nullptr, neW[1], P(NODEE + 4), h1);
  gemm_mlp<1, 8><<<grid(N), 256, shmem(128, 1, 8), stream>>>(
      N, 128, 1, h1, nullptr, nullptr, nullptr, nullptr, nullptr, neW[2],
      P(NODEE + 5), 128, P(NODEE + 7), P(NODEE + 6), vF, vB, nullptr, nullptr,
      nullptr, nullptr);
  // mesh encoder -> e0
  gemmRelu(E0, 32, 1, xm, nullptr, nullptr, nullptr, nullptr, nullptr, meW[0], P(MESHE + 3), h0);
  gemmRelu(E0, 128, 1, h0, nullptr, nullptr, nullptr, nullptr, nullptr, meW[1], P(MESHE + 4), h1);
  gemm_mlp<1, 8><<<grid(E0), 256, shmem(128, 1, 8), stream>>>(
      E0, 128, 1, h1, nullptr, nullptr, nullptr, nullptr, nullptr, meW[2],
      P(MESHE + 5), 128, P(MESHE + 7), P(MESHE + 6), e0F, e0B, nullptr, nullptr,
      nullptr, nullptr);
  // world encoder -> e1
  if (EW > 0) {
    gemmRelu(EW, 32, 1, xw, nullptr, nullptr, nullptr, nullptr, nullptr, weW[0], P(WORLDE + 3), h0);
    gemmRelu(EW, 128, 1, h0, nullptr, nullptr, nullptr, nullptr, nullptr, weW[1], P(WORLDE + 4), h1);
    gemm_mlp<1, 8><<<grid(EW), 256, shmem(128, 1, 8), stream>>>(
        EW, 128, 1, h1, nullptr, nullptr, nullptr, nullptr, nullptr, weW[2],
        P(WORLDE + 5), 128, P(WORLDE + 7), P(WORLDE + 6), e1F, e1B, nullptr,
        nullptr, nullptr, nullptr);
  }

  // ---- message passing ----
  for (int i = 0; i < MP_STEPS; ++i) {
    hipMemsetAsync(ag0, 0, (size_t)N * LATENT * 4, stream);
    hipMemsetAsync(ag1, 0, (size_t)N * LATENT * 4, stream);

    // mesh-edge MLP: concat([e0, v[srcs], v[dsts]]) -> e0n; fused residual+scatter
    gemmRelu(E0, 128, 3, e0B, vB, vB, nullptr, srcs, dsts, bW[i][0][0], BLK(i, 0, 3), h0);
    gemmRelu(E0, 128, 1, h0, nullptr, nullptr, nullptr, nullptr, nullptr, bW[i][0][1], BLK(i, 0, 4), h1);
    gemm_mlp<3, 8><<<grid(E0), 256, shmem(128, 1, 8), stream>>>(
        E0, 128, 1, h1, nullptr, nullptr, nullptr, nullptr, nullptr,
        bW[i][0][2], BLK(i, 0, 5), 128, BLK(i, 0, 7), BLK(i, 0, 6), e0F, e0B,
        dsts, ag0, nullptr, nullptr);

    // world-edge MLP
    if (EW > 0) {
      gemmRelu(EW, 128, 3, e1B, vB, vB, nullptr, wsrcs, wdsts, bW[i][1][0], BLK(i, 1, 3), h0);
      gemmRelu(EW, 128, 1, h0, nullptr, nullptr, nullptr, nullptr, nullptr, bW[i][1][1], BLK(i, 1, 4), h1);
      gemm_mlp<3, 8><<<grid(EW), 256, shmem(128, 1, 8), stream>>>(
          EW, 128, 1, h1, nullptr, nullptr, nullptr, nullptr, nullptr,
          bW[i][1][2], BLK(i, 1, 5), 128, BLK(i, 1, 7), BLK(i, 1, 6), e1F, e1B,
          wdsts, ag1, nullptr, nullptr);
    }

    cvt(ag0, ag0B, (long)N * LATENT);
    cvt(ag1, ag1B, (long)N * LATENT);

    // node MLP: concat([v, agg0, agg1]) -> vn; fused residual
    gemmRelu(N, 128, 3, vB, ag0B, ag1B, nullptr, nullptr, nullptr, bW[i][2][0], BLK(i, 2, 3), h0);
    gemmRelu(N, 128, 1, h0, nullptr, nullptr, nullptr, nullptr, nullptr, bW[i][2][1], BLK(i, 2, 4), h1);
    gemm_mlp<2, 8><<<grid(N), 256, shmem(128, 1, 8), stream>>>(
        N, 128, 1, h1, nullptr, nullptr, nullptr, nullptr, nullptr,
        bW[i][2][2], BLK(i, 2, 5), 128, BLK(i, 2, 7), BLK(i, 2, 6), vF, vB,
        nullptr, nullptr, nullptr, nullptr);
  }

  // ---- decoder ----
  gemmRelu(N, 128, 1, vB, nullptr, nullptr, nullptr, nullptr, nullptr, dcW[0], P(DEC + 3), h0);
  gemmRelu(N, 128, 1, h0, nullptr, nullptr, nullptr, nullptr, nullptr, dcW[1], P(DEC + 4), h1);
  gemm_mlp<4, 1><<<grid(N), 256, shmem(128, 1, 1), stream>>>(
      N, 128, 1, h1, nullptr, nullptr, nullptr, nullptr, nullptr, dcW[2],
      P(DEC + 5), 3, nullptr, nullptr, (float*)d_out, nullptr, nullptr, nullptr,
      P(OUTN + 1) /*std*/, P(OUTN) /*mean*/);
}
